// MiniCrossAttentionModule_38130719654447
// MI455X (gfx1250) — matrix operand order repaired and checked by reading, not compiled, not run on hardware
//
#include <hip/hip_runtime.h>
#include <hip/hip_bf16.h>
#include <math.h>

typedef __attribute__((ext_vector_type(16))) __bf16 v16bf;
typedef __attribute__((ext_vector_type(8)))  float  v8f;
typedef __attribute__((ext_vector_type(4)))  int    v4i;

#define E_DIM 512
#define T_DIM 1024
#define S_DIM 2048
#define N_B   8
#define S_BLK 64
#define KVSTR  520     // kv_lds row stride (bf16): 512 + 8 pad
#define KVTSTR 72      // kvT_lds row stride (bf16): 64 + 8 pad
#define PSTR   68      // raw partial row stride (f32): 64 + 4 pad
#define PBSTR  72      // p_bf row stride (bf16): 64 + 8 pad
#define NEG_BIG (-1.0e30f)
#define SCALE 0.04419417382415922f   // 1/sqrt(512)

union Frag16 { v16bf v; uint4 q[2]; };

// Optional CDNA5 async global->LDS staging (probe via __has_builtin)
#if defined(__has_builtin)
#if __has_builtin(__builtin_amdgcn_global_load_async_to_lds_b128) && \
    __has_builtin(__builtin_amdgcn_s_wait_asynccnt)
#define HAVE_ASYNC_LDS 1
#endif
#endif
#ifndef HAVE_ASYNC_LDS
#define HAVE_ASYNC_LDS 0
#endif

#if HAVE_ASYNC_LDS
// prototype (from clang diagnostic): arg0 = global v4i* (AS1), arg1 = LDS v4i* (AS3)
#define GLB_PTR(p) ((__attribute__((address_space(1))) v4i*)(p))
#define LDS_PTR(p) ((__attribute__((address_space(3))) v4i*)(p))
#endif

// ---------------- Kernel 1: fused LayerNorm (target & source) -> bf16 ----------------
__global__ void ln_to_bf16_kernel(const float* __restrict__ target,
                                  const float* __restrict__ source,
                                  const float* __restrict__ wt, const float* __restrict__ bt,
                                  const float* __restrict__ ws, const float* __restrict__ bs,
                                  __bf16* __restrict__ q_bf, __bf16* __restrict__ kv_bf) {
    __shared__ float rs[256];
    __shared__ float rq[256];
    const int row = blockIdx.x;
    const int tid = threadIdx.x;
    const float* x; const float* w; const float* b; __bf16* dst;
    if (row < N_B * T_DIM) {
        x = target + (size_t)row * E_DIM; w = wt; b = bt;
        dst = q_bf + (size_t)row * E_DIM;
    } else {
        const int rr = row - N_B * T_DIM;
        x = source + (size_t)rr * E_DIM; w = ws; b = bs;
        dst = kv_bf + (size_t)rr * E_DIM;
    }
    const float x0 = x[tid];
    const float x1 = x[tid + 256];
    rs[tid] = x0 + x1;
    rq[tid] = x0 * x0 + x1 * x1;
    __syncthreads();
    for (int off = 128; off > 0; off >>= 1) {
        if (tid < off) { rs[tid] += rs[tid + off]; rq[tid] += rq[tid + off]; }
        __syncthreads();
    }
    const float mu   = rs[0] * (1.0f / E_DIM);
    const float var  = rq[0] * (1.0f / E_DIM) - mu * mu;
    const float rstd = rsqrtf(var + 1e-5f);
    dst[tid]       = (__bf16)((x0 - mu) * rstd * w[tid]       + b[tid]);
    dst[tid + 256] = (__bf16)((x1 - mu) * rstd * w[tid + 256] + b[tid + 256]);
}

// ---------------- Kernel 1b: kv (n,s,e) -> kvT (n,e,s), LDS-tiled 32x32 ----------------
__global__ void transpose_kv_kernel(const __bf16* __restrict__ kv_bf,
                                    __bf16* __restrict__ kvT) {
    __shared__ __bf16 tile[32][34];
    const int n  = blockIdx.z;
    const int s0 = blockIdx.x * 32;
    const int e0 = blockIdx.y * 32;
    const int tc = threadIdx.x & 31;
    const int tr = threadIdx.x >> 5;     // 0..7
    #pragma unroll
    for (int i = 0; i < 4; ++i) {
        const int r = tr + i * 8;
        tile[r][tc] = kv_bf[((size_t)n * S_DIM + s0 + r) * E_DIM + e0 + tc];
    }
    __syncthreads();
    #pragma unroll
    for (int i = 0; i < 4; ++i) {
        const int r = tr + i * 8;
        kvT[((size_t)n * E_DIM + e0 + r) * S_DIM + s0 + tc] = tile[tc][r];
    }
}

// ---------------- Kernel 2: flash cross-attention, bf16 WMMA ----------------
__global__ __launch_bounds__(256, 1)
void flash_attn_kernel(const __bf16* __restrict__ q_bf,
                       const __bf16* __restrict__ kv_bf,
                       const __bf16* __restrict__ kvT_g,
                       const unsigned char* __restrict__ mask,
                       float* __restrict__ out) {
    extern __shared__ char smem_raw[];
    __bf16* kv_lds  = (__bf16*)smem_raw;                  // [S_BLK][KVSTR]   (row-major, K=E)
    __bf16* kvT_lds = kv_lds + S_BLK * KVSTR;             // [E_DIM][KVTSTR]  (S-major, K=S)
    __bf16* q_lds   = kvT_lds + E_DIM * KVTSTR;           // [16][KVSTR], dead after preamble
    // overlap zone (reuses q_lds space once Q fragments are hoisted to registers):
    float*  p1      = (float*)q_lds;                      // [16][PSTR] raw partial (khalf 0)
    float*  p2      = p1 + 16 * PSTR;                     // [16][PSTR] raw partial (khalf 1)
    float*  mb_lds  = p2 + 16 * PSTR;                     // [64] additive mask bias (0 or -1e30)
    float*  m_lds   = mb_lds + 64;                        // [16] running max
    float*  l_lds   = m_lds + 16;                         // [16] running sum
    float*  al_lds  = l_lds + 16;                         // [16] rescale alpha
    __bf16* p_bf    = (__bf16*)(al_lds + 16);             // [16][PBSTR] P tile in bf16

    const int tid   = threadIdx.x;
    const int wave  = tid >> 5;
    const int lane  = tid & 31;
    const int lm    = lane & 15;
    const int lh    = lane >> 4;
    const int j4    = wave & 3;       // score column tile selector
    const int khalf = wave >> 2;      // split-K half for GEMM1
    const int n     = blockIdx.y;
    const int t0    = blockIdx.x * 16;
    const unsigned char* maskp = mask + (size_t)n * S_DIM;

    // ---- preamble: stage Q tile, hoist this wave's A fragments into registers ----
    const __bf16* qrow = q_bf + ((size_t)n * T_DIM + t0) * E_DIM;
    #pragma unroll
    for (int it = 0; it < 2; ++it) {
        const int c = it * 4096 + tid * 16;
        const int r = c >> 9, col = c & 511;
        *(uint4*)&q_lds[r * KVSTR + col] = *(const uint4*)&qrow[r * E_DIM + col];
    }
    __syncthreads();
    Frag16 qf[8];
    {
        const __bf16* arow = &q_lds[lm * KVSTR];
        #pragma unroll
        for (int ekk = 0; ekk < 8; ++ekk) {
            const int e0 = (khalf * 8 + ekk) * 32;
            qf[ekk].q[0] = *(const uint4*)&arow[e0 + lh * 8];
            qf[ekk].q[1] = *(const uint4*)&arow[e0 + 16 + lh * 8];
        }
    }
    __syncthreads();   // everyone done reading q_lds; overlap zone may now be written
    if (tid < 16) { m_lds[tid] = NEG_BIG; l_lds[tid] = 0.0f; al_lds[tid] = 1.0f; }

    v8f acc[4];
    #pragma unroll
    for (int i = 0; i < 4; ++i)
        #pragma unroll
        for (int r = 0; r < 8; ++r) acc[i][r] = 0.0f;

    for (int s0 = 0; s0 < S_DIM; s0 += S_BLK) {
        __syncthreads();   // previous iteration's readers of kv/kvT/p done (also stats init)

        const __bf16* kvrow  = kv_bf + ((size_t)n * S_DIM + s0) * E_DIM;   // 64 x 512
        const __bf16* kvTrow = kvT_g + (size_t)n * E_DIM * S_DIM + s0;     // 512-row x 64 slice
        if (s0 + S_BLK < S_DIM) {
            __builtin_prefetch((const char*)(kvrow + S_BLK * E_DIM) + tid * 256, 0, 1);
            __builtin_prefetch((const char*)(kvrow + S_BLK * E_DIM) + tid * 256 + 128, 0, 1);
            __builtin_prefetch(&kvTrow[(size_t)(tid * 2) * S_DIM + S_BLK], 0, 1);
        }
        // stage kv (row-major) and kvT (S-major) tiles into LDS
        #pragma unroll
        for (int it = 0; it < 8; ++it) {
            const int c = it * 4096 + tid * 16;
            const int r = c >> 9, col = c & 511;
#if HAVE_ASYNC_LDS
            __builtin_amdgcn_global_load_async_to_lds_b128(
                GLB_PTR(&kvrow[r * E_DIM + col]), LDS_PTR(&kv_lds[r * KVSTR + col]), 0, 0);
#else
            *(uint4*)&kv_lds[r * KVSTR + col] = *(const uint4*)&kvrow[r * E_DIM + col];
#endif
        }
        #pragma unroll
        for (int it = 0; it < 8; ++it) {
            const int c = it * 4096 + tid * 16;
            const int r = c >> 6, col = c & 63;
#if HAVE_ASYNC_LDS
            __builtin_amdgcn_global_load_async_to_lds_b128(
                GLB_PTR(&kvTrow[(size_t)r * S_DIM + col]), LDS_PTR(&kvT_lds[r * KVTSTR + col]), 0, 0);
#else
            *(uint4*)&kvT_lds[r * KVTSTR + col] = *(const uint4*)&kvTrow[(size_t)r * S_DIM + col];
#endif
        }
        if (tid < S_BLK) mb_lds[tid] = maskp[s0 + tid] ? 0.0f : NEG_BIG;
#if HAVE_ASYNC_LDS
        __builtin_amdgcn_s_wait_asynccnt(0);
#endif
        __syncthreads();

        // ---- GEMM1 (all 8 waves, split-K over E): raw partial scores(16x64) ----
        v8f sacc;
        #pragma unroll
        for (int r = 0; r < 8; ++r) sacc[r] = 0.0f;
        const __bf16* brow = &kv_lds[(j4 * 16 + lm) * KVSTR + khalf * 256];
        #pragma unroll
        for (int ekk = 0; ekk < 8; ++ekk) {
            const int e0 = ekk * 32;
            Frag16 b;
            b.q[0] = *(const uint4*)&brow[e0 + lh * 8];
            b.q[1] = *(const uint4*)&brow[e0 + 16 + lh * 8];
            sacc = __builtin_amdgcn_wmma_f32_16x16x32_bf16(
                false, qf[ekk].v, false, b.v, (short)0, sacc, false, false);
        }
        {
            float* pd = (khalf == 0) ? p1 : p2;
            const int col = j4 * 16 + lm;
            #pragma unroll
            for (int r = 0; r < 8; ++r)
                pd[(r + 8 * lh) * PSTR + col] = sacc[r];
        }
        __syncthreads();

        // ---- parallel online softmax: wave w owns rows 2w, 2w+1; lane handles 4 cols ----
        // writes P directly as bf16 so GEMM2 needs no conversions
        {
            const int r = wave * 2 + lh;
            const float* p1r = p1 + r * PSTR;
            const float* p2r = p2 + r * PSTR;
            __bf16* pbr = p_bf + r * PBSTR;
            float v[4];
            #pragma unroll
            for (int i = 0; i < 4; ++i) {
                const int c = lm + 16 * i;
                v[i] = (p1r[c] + p2r[c]) * SCALE + mb_lds[c];
            }
            float bmax = fmaxf(fmaxf(v[0], v[1]), fmaxf(v[2], v[3]));
            #pragma unroll
            for (int mdist = 1; mdist < 16; mdist <<= 1)
                bmax = fmaxf(bmax, __shfl_xor(bmax, mdist, 32));
            const float mold = m_lds[r];
            const float lold = l_lds[r];
            const float mnew = fmaxf(mold, bmax);
            const float alpha = __expf(mold - mnew);
            float s = 0.0f;
            #pragma unroll
            for (int i = 0; i < 4; ++i) {
                const float p = __expf(v[i] - mnew);
                pbr[lm + 16 * i] = (__bf16)p;
                s += p;
            }
            #pragma unroll
            for (int mdist = 1; mdist < 16; mdist <<= 1)
                s += __shfl_xor(s, mdist, 32);
            if (lm == 0) {
                m_lds[r]  = mnew;
                l_lds[r]  = lold * alpha + s;
                al_lds[r] = alpha;
            }
        }
        __syncthreads();

        // ---- rescale accumulators, then GEMM2: O_slice += P(16x64) @ KV(64 x 64slice) ----
        #pragma unroll
        for (int r = 0; r < 8; ++r) {
            const float al = al_lds[r + 8 * lh];
            #pragma unroll
            for (int i = 0; i < 4; ++i) acc[i][r] *= al;
        }
        const __bf16* prow = &p_bf[lm * PBSTR];
        #pragma unroll
        for (int nt = 0; nt < 4; ++nt) {
            const int ec = wave * 64 + nt * 16;   // this wave's E-column slice
            const __bf16* btrow = &kvT_lds[(ec + lm) * KVTSTR];
            #pragma unroll
            for (int kc = 0; kc < 2; ++kc) {
                const int ks = kc * 32;
                Frag16 a, b;
                a.q[0] = *(const uint4*)&prow[ks + lh * 8];
                a.q[1] = *(const uint4*)&prow[ks + 16 + lh * 8];
                b.q[0] = *(const uint4*)&btrow[ks + lh * 8];
                b.q[1] = *(const uint4*)&btrow[ks + 16 + lh * 8];
                acc[nt] = __builtin_amdgcn_wmma_f32_16x16x32_bf16(
                    false, a.v, false, b.v, (short)0, acc[nt], false, false);
            }
        }
    }
    __syncthreads();

    // ---- finalize: divide by l, store ----
    float inv[8];
    #pragma unroll
    for (int r = 0; r < 8; ++r) inv[r] = 1.0f / l_lds[r + 8 * lh];
    #pragma unroll
    for (int nt = 0; nt < 4; ++nt) {
        const int col = wave * 64 + nt * 16 + lm;
        #pragma unroll
        for (int r = 0; r < 8; ++r) {
            const int row = t0 + r + 8 * lh;
            out[((size_t)n * T_DIM + row) * E_DIM + col] = acc[nt][r] * inv[r];
        }
    }
}

extern "C" void kernel_launch(void* const* d_in, const int* in_sizes, int n_in,
                              void* d_out, int out_size, void* d_ws, size_t ws_size,
                              hipStream_t stream) {
    (void)in_sizes; (void)n_in; (void)out_size; (void)ws_size;
    const float* target = (const float*)d_in[0];
    const float* source = (const float*)d_in[1];
    const float* ln_t_w = (const float*)d_in[2];
    const float* ln_t_b = (const float*)d_in[3];
    const float* ln_s_w = (const float*)d_in[4];
    const float* ln_s_b = (const float*)d_in[5];
    const unsigned char* mask = (const unsigned char*)d_in[6];
    float* out = (float*)d_out;

    __bf16* q_bf  = (__bf16*)d_ws;                        // 8 MB
    __bf16* kv_bf = q_bf + (size_t)N_B * T_DIM * E_DIM;   // 16 MB
    __bf16* kvT   = kv_bf + (size_t)N_B * S_DIM * E_DIM;  // 16 MB

    // 1) LayerNorm target+source -> bf16
    ln_to_bf16_kernel<<<N_B * T_DIM + N_B * S_DIM, 256, 0, stream>>>(
        target, source, ln_t_w, ln_t_b, ln_s_w, ln_s_b, q_bf, kv_bf);

    // 1b) kv -> kvT (S-major copy for GEMM2 B fragments)
    dim3 tgrid(S_DIM / 32, E_DIM / 32, N_B);
    transpose_kv_kernel<<<tgrid, 256, 0, stream>>>(kv_bf, kvT);

    // 2) fused flash attention
    constexpr int SMEM_BYTES =
        (S_BLK * KVSTR + E_DIM * KVTSTR + 16 * KVSTR) * 2;   // 156,928 B
    (void)hipFuncSetAttribute((const void*)flash_attn_kernel,
                              hipFuncAttributeMaxDynamicSharedMemorySize, SMEM_BYTES);
    dim3 grid(T_DIM / 16, N_B);
    flash_attn_kernel<<<grid, 256, SMEM_BYTES, stream>>>(q_bf, kv_bf, kvT, mask, out);
}
